// Pair_loss_80496277062328
// MI455X (gfx1250) — matrix operand order repaired and checked by reading, not compiled, not run on hardware
//
#include <hip/hip_runtime.h>
#include <hip/hip_bf16.h>

typedef _Float16 v16h __attribute__((ext_vector_type(16)));
typedef _Float16 v8h  __attribute__((ext_vector_type(8)));
typedef float    v8f  __attribute__((ext_vector_type(8)));

#define NN 8192
#define MM 8192
#define DD 256
#define EPS 1e-7f

union F16Frag {
    v16h v;
    v8h  h[2];
};

// ---------------------------------------------------------------------------
// Kernel 1: convert fp32 probability rows -> f16 in workspace, zero d_out.
// Each thread converts 4 elements of prob_1 and 4 elements of prob_2.
// ---------------------------------------------------------------------------
__global__ void cvt_f32_to_f16_kernel(const float* __restrict__ p1,
                                      const float* __restrict__ p2,
                                      _Float16* __restrict__ h1,
                                      _Float16* __restrict__ h2,
                                      float* __restrict__ out) {
    if (blockIdx.x == 0 && threadIdx.x == 0) out[0] = 0.0f;
    size_t i = ((size_t)blockIdx.x * blockDim.x + threadIdx.x) * 4;
    const size_t ND = (size_t)NN * DD;
    if (i < ND) {
        float4 a = *(const float4*)(p1 + i);
        h1[i + 0] = (_Float16)a.x; h1[i + 1] = (_Float16)a.y;
        h1[i + 2] = (_Float16)a.z; h1[i + 3] = (_Float16)a.w;
        float4 b = *(const float4*)(p2 + i);
        h2[i + 0] = (_Float16)b.x; h2[i + 1] = (_Float16)b.y;
        h2[i + 2] = (_Float16)b.z; h2[i + 3] = (_Float16)b.w;
    }
}

// ---------------------------------------------------------------------------
// Kernel 2: fused GEMM (WMMA f16 -> f32) + pairwise BCE loss + reduction.
//
// Block tile: 128 (M, prob_2 rows) x 128 (N, prob_1 rows).
// 256 threads = 8 waves (wave32). Waves arranged 4 (M) x 2 (N):
//   wave owns a 32x64 region = 2 x 4 tiles of 16x16.
//
// WMMA fragment layouts (per CDNA5 ISA, 16-bit A 16x32 / B 32x16):
//   A: lane L holds row (L&15); halves e=0..7 at K = kk + (L>>4)*8 + e,
//                              halves e=8..15 at K = kk + 16 + (L>>4)*8 + (e-8)
//      -> two contiguous 8-half (16B) loads at k and k+16.
//   B: lane L holds column (L&15); same K mapping as A
//      -> two contiguous 8-half loads at k and k+16.
//   C/D f32: VGPR j: lanes 0-15 -> row j, lanes 16-31 -> row j+8; col = L&15.
// ---------------------------------------------------------------------------
__global__ void __launch_bounds__(256)
pair_loss_wmma_kernel(const _Float16* __restrict__ hA,   // prob_2 f16 [MM][DD]
                      const _Float16* __restrict__ hB,   // prob_1 f16 [NN][DD]
                      const int* __restrict__ gt1,
                      const int* __restrict__ gt2,
                      const int* __restrict__ mask1,
                      const int* __restrict__ mask2,
                      float* __restrict__ out) {
    __shared__ int   s_cls1[128];   // per-column class (prob_1 side), -1 if masked out
    __shared__ int   s_cls2[128];   // per-row class (prob_2 side), -2 if masked out
    __shared__ float s_red[8];

    const int tid    = threadIdx.x;
    const int lane   = tid & 31;
    const int wave   = tid >> 5;
    const int waveM  = wave & 3;    // 0..3 -> 32-row strip
    const int waveN  = wave >> 2;   // 0..1 -> 64-col strip
    const int blockM = blockIdx.y;  // 0..63
    const int blockN = blockIdx.x;  // 0..63

    // Preload fused class/mask values for this 128x128 tile.
    if (tid < 128) {
        int c = blockN * 128 + tid;
        s_cls1[tid] = (mask1[c] != 0) ? gt1[c] : -1;
        int r = blockM * 128 + tid;
        s_cls2[tid] = (mask2[r] != 0) ? gt2[r] : -2;
    }
    __syncthreads();

    const int l15   = lane & 15;
    const int lhi   = lane >> 4;          // 0 or 1
    const int rowA0 = blockM * 128 + waveM * 32 + l15;   // A rows for im=0 tile
    const int colB0 = blockN * 128 + waveN * 64 + l15;   // B cols for in=0 tile
    const int kOffA = lhi << 3;           // +0 / +8
    const int kOffB = lhi << 3;           // +0 / +8

    v8f acc[2][4];
    #pragma unroll
    for (int im = 0; im < 2; ++im)
        #pragma unroll
        for (int in = 0; in < 4; ++in)
            acc[im][in] = (v8f){0.f, 0.f, 0.f, 0.f, 0.f, 0.f, 0.f, 0.f};

    for (int kk = 0; kk < DD; kk += 32) {
        F16Frag a[2], b[4];
        #pragma unroll
        for (int im = 0; im < 2; ++im) {
            const _Float16* pa = hA + (size_t)(rowA0 + im * 16) * DD + kk + kOffA;
            a[im].h[0] = *(const v8h*)(pa);
            a[im].h[1] = *(const v8h*)(pa + 16);
            // WGP-scope prefetch of the next K-slab (locality 3 -> SCOPE 0:
            // fills all cache levels, incl. the near WGP cache).
            if (kk + 32 < DD) __builtin_prefetch(pa + 32, 0, 3);
        }
        #pragma unroll
        for (int in = 0; in < 4; ++in) {
            const _Float16* pb = hB + (size_t)(colB0 + in * 16) * DD + kk + kOffB;
            b[in].h[0] = *(const v8h*)(pb);
            b[in].h[1] = *(const v8h*)(pb + 16);
            if (kk + 32 < DD) __builtin_prefetch(pb + 32, 0, 3);
        }
        #pragma unroll
        for (int im = 0; im < 2; ++im)
            #pragma unroll
            for (int in = 0; in < 4; ++in)
                acc[im][in] = __builtin_amdgcn_wmma_f32_16x16x32_f16(
                    false, a[im].v, false, b[in].v,
                    (short)0, acc[im][in], false, false);
    }

    // Fused elementwise loss: t in {0,1} -> single log per element.
    float lsum = 0.0f;
    #pragma unroll
    for (int im = 0; im < 2; ++im) {
        #pragma unroll
        for (int in = 0; in < 4; ++in) {
            const int colL = waveN * 64 + in * 16 + l15;
            const int cls1 = s_cls1[colL];
            #pragma unroll
            for (int j = 0; j < 8; ++j) {
                const int rowL = waveM * 32 + im * 16 + j + (lhi << 3);
                const bool t   = (s_cls2[rowL] == cls1);
                const float p  = acc[im][in][j];
                const float arg = t ? (p + EPS) : (1.0f - p + EPS);
                lsum += __logf(arg);
            }
        }
    }

    // Wave reduction (wave32), then cross-wave via LDS, then one atomic.
    #pragma unroll
    for (int off = 16; off > 0; off >>= 1)
        lsum += __shfl_xor(lsum, off, 32);
    if (lane == 0) s_red[wave] = lsum;
    __syncthreads();
    if (tid == 0) {
        float bsum = 0.0f;
        #pragma unroll
        for (int w = 0; w < 8; ++w) bsum += s_red[w];
        const float scale = -1.0f / ((float)MM * (float)NN);
        atomicAdd(out, bsum * scale);
    }
}

// ---------------------------------------------------------------------------
// Host launcher. Input order (setup_inputs dict): gt1, gt2, mask1, mask2,
// prob_1, prob_2. Output: single float (mean loss).
// ---------------------------------------------------------------------------
extern "C" void kernel_launch(void* const* d_in, const int* in_sizes, int n_in,
                              void* d_out, int out_size, void* d_ws, size_t ws_size,
                              hipStream_t stream) {
    const int*   gt1   = (const int*)d_in[0];
    const int*   gt2   = (const int*)d_in[1];
    const int*   mask1 = (const int*)d_in[2];
    const int*   mask2 = (const int*)d_in[3];
    const float* prob1 = (const float*)d_in[4];
    const float* prob2 = (const float*)d_in[5];
    float* out = (float*)d_out;

    // Workspace layout: [MM*DD halves] prob_2 f16 (A side), then
    //                   [NN*DD halves] prob_1 f16 (B side).
    _Float16* hA = (_Float16*)d_ws;                 // prob_2 -> A (M side)
    _Float16* hB = hA + (size_t)MM * DD;            // prob_1 -> B (N side)

    // 1) fp32 -> f16 conversion (and d_out zeroing). 4 elems/thread/array.
    {
        const size_t ND = (size_t)NN * DD;
        int blocks = (int)(ND / 4 / 256);           // 2048
        cvt_f32_to_f16_kernel<<<blocks, 256, 0, stream>>>(prob1, prob2, hB, hA, out);
    }

    // 2) fused WMMA GEMM + BCE loss + mean reduction.
    dim3 grid(NN / 128, MM / 128);                  // 64 x 64
    pair_loss_wmma_kernel<<<grid, 256, 0, stream>>>(hA, hB, gt1, gt2, mask1, mask2, out);
}
